// ScalarPotentialLMSARFMass_48180943127155
// MI455X (gfx1250) — matrix operand order repaired and checked by reading, not compiled, not run on hardware
//
#include <hip/hip_runtime.h>
#include <hip/hip_bf16.h>
#include <math.h>

// ---- problem constants (from reference) ----
#define B_ 8
#define T_ 512
#define D_ 128
#define H_ 512
#define V_ 50257
#define L_ 8
#define M_ (B_ * T_)          // 4096 tokens
#define NC_ 16                // scan chunks over T
#define CH_ (T_ / NC_)        // 32

typedef __bf16 bf16;
typedef __bf16 bf16x16 __attribute__((ext_vector_type(16)));
typedef float  v8f     __attribute__((ext_vector_type(8)));
typedef int    v4i_    __attribute__((vector_size(16)));   // matches async-LDS builtin pointee

#define AS1 __attribute__((address_space(1)))
#define AS3 __attribute__((address_space(3)))

#if defined(__has_builtin)
#  if __has_builtin(__builtin_amdgcn_global_load_async_to_lds_b128)
#    define USE_ASYNC_LDS 1
#  else
#    define USE_ASYNC_LDS 0
#  endif
#else
#  define USE_ASYNC_LDS 0
#endif

__device__ __forceinline__ void async_lds_b128x2(const bf16* src, bf16* dst) {
#if USE_ASYNC_LDS
  // 32 contiguous bytes per thread: two async b128 transfers (IOFFSET applies
  // to both global and LDS addresses per ISA pseudocode).
  __builtin_amdgcn_global_load_async_to_lds_b128((AS1 v4i_*)src, (AS3 v4i_*)dst, 0, 0);
  __builtin_amdgcn_global_load_async_to_lds_b128((AS1 v4i_*)src, (AS3 v4i_*)dst, 16, 0);
#else
  const uint4* s4 = (const uint4*)src;
  *(uint4*)dst       = s4[0];
  *(uint4*)(dst + 8) = s4[1];
#endif
}

__device__ __forceinline__ void wait_asynccnt0() {
#if defined(__has_builtin)
#  if __has_builtin(__builtin_amdgcn_s_wait_asynccnt)
  __builtin_amdgcn_s_wait_asynccnt(0);
  return;
#  else
  asm volatile("s_wait_asynccnt 0" ::: "memory");
  return;
#  endif
#else
  asm volatile("s_wait_asynccnt 0" ::: "memory");
#endif
}

union FragBf { bf16x16 v; uint4 u[2]; };

__device__ __forceinline__ float geluf(float x) {
  return 0.5f * x * (1.0f + erff(x * 0.7071067811865476f));
}
__device__ __forceinline__ float gelugf(float x) {
  float cdf = 0.5f * (1.0f + erff(x * 0.7071067811865476f));
  return cdf + x * 0.3989422804014327f * expf(-0.5f * x * x);
}
__device__ __forceinline__ float softplusf(float x) {
  return (x > 20.0f) ? x : log1pf(expf(x));
}

// ------------------------------------------------------------------
// Embedding gather: h[bt,d] = E[x[bt],d] + P[t,d]; v = 0
// ------------------------------------------------------------------
__global__ void embed_kernel(const int* __restrict__ x, const float* __restrict__ E,
                             const float* __restrict__ P, float* __restrict__ h,
                             float* __restrict__ v) {
  int bt = blockIdx.x;
  int d  = threadIdx.x;
  int t  = bt & (T_ - 1);
  int tok = x[bt];
  size_t idx = (size_t)bt * D_ + d;
  h[idx] = E[(size_t)tok * D_ + d] + P[(size_t)t * D_ + d];
  v[idx] = 0.0f;
}

// ------------------------------------------------------------------
// f32 -> bf16 conversion (straight and transposed weight copies)
// ------------------------------------------------------------------
__global__ void cvt_bf16_kernel(const float* __restrict__ in, bf16* __restrict__ out, int n) {
  int i = blockIdx.x * blockDim.x + threadIdx.x;
  if (i < n) out[i] = (bf16)in[i];
}
// out[c*R + r] = in[r*C + c]   (in: R x C row-major -> out: C x R row-major)
__global__ void cvt_t_bf16_kernel(const float* __restrict__ in, bf16* __restrict__ out,
                                  int R, int C) {
  int i = blockIdx.x * blockDim.x + threadIdx.x;
  if (i < R * C) {
    int r = i / C, c = i - r * C;
    out[(size_t)c * R + r] = (bf16)in[i];
  }
}

// ------------------------------------------------------------------
// Causal cumulative mean -> z packing, two-phase chunked scan over T.
// z[bt, 0:D] = bf16(cummean_t(h)), z[bt, D:2D] = bf16(h)
// ------------------------------------------------------------------
__global__ void chunk_sum_kernel(const float* __restrict__ h, float* __restrict__ csum) {
  int b = blockIdx.x, c = blockIdx.y, d = threadIdx.x;
  float s = 0.0f;
  for (int t = c * CH_; t < (c + 1) * CH_; ++t)
    s += h[((size_t)(b * T_ + t)) * D_ + d];
  csum[((size_t)b * NC_ + c) * D_ + d] = s;
}
__global__ void pack_kernel(const float* __restrict__ h, const float* __restrict__ csum,
                            bf16* __restrict__ z) {
  int b = blockIdx.x, c = blockIdx.y, d = threadIdx.x;
  float acc = 0.0f;
  for (int cc = 0; cc < c; ++cc)
    acc += csum[((size_t)b * NC_ + cc) * D_ + d];
  for (int t = c * CH_; t < (c + 1) * CH_; ++t) {
    float hv = h[((size_t)(b * T_ + t)) * D_ + d];
    acc += hv;
    bf16* zp = z + ((size_t)(b * T_ + t)) * (2 * D_);
    zp[d]      = (bf16)(acc / (float)(t + 1));
    zp[D_ + d] = (bf16)hv;
  }
}

// ------------------------------------------------------------------
// bf16 WMMA GEMM: 256 threads (8 wave32), 128x128 block tile, K-step 32.
// Each wave computes 32x64 (2x4 fragments -> 8 wmma per K-step).
//   C = A(MxK) * B(KxN), B staged via transposed ("BT") addressing:
//     B element (k,n) = Bsrc[n*ldb + k]
// bf16 staging uses GLOBAL_LOAD_ASYNC_TO_LDS_B128 (ASYNCcnt +
// s_wait_asynccnt); fp32-source paths bounce via VGPRs with conversion.
// Epilogues:
//   EPI 0: fp32 store with N-guard                -> Cout (logits)
//   EPI 1: x=C+bias; store gelu(x), gelu'(x)      -> OutBf, Gout
//   EPI 2: x=C+bias; store W3[col]*gelu'(x)       -> OutBf (da2 seed)
//   EPI 3: store C * Gin                          -> OutBf (backward chain)
//   EPI 4: grad=C; v=(v-grad/m)/(1+gamma); h+=v   -> Hbuf/Vbuf update
// ------------------------------------------------------------------
#define TILE 128
#define KSTEP 32
#define LDSS 48   // LDS row stride (elements); 96B rows keep 16B alignment

template <int EPI, bool AF32, bool BF32, bool NGUARD>
__global__ __launch_bounds__(256)
void gemm_wmma(const void* __restrict__ Aptr, const void* __restrict__ Bptr,
               int Mdim, int Ndim, int Kdim, int lda, int ldb, int ldc,
               const float* __restrict__ bias, const float* __restrict__ W3,
               bf16* __restrict__ OutBf, const bf16* __restrict__ Gin,
               bf16* __restrict__ Gout, float* __restrict__ Cout,
               float* __restrict__ Hbuf, float* __restrict__ Vbuf,
               const float* __restrict__ rawm, const float* __restrict__ rawg) {
  __shared__ __align__(16) bf16 As[TILE * LDSS];
  __shared__ __align__(16) bf16 Bs[TILE * LDSS];

  const int tid  = threadIdx.x;
  const int lane = tid & 31;
  const int wave = tid >> 5;          // 0..7
  const int wm = (wave & 3) * 32;     // 4 row groups of 32
  const int wn = (wave >> 2) * 64;    // 2 col groups of 64
  const int m0 = blockIdx.y * TILE;
  const int n0 = blockIdx.x * TILE;

  const v8f vzero = {0.f, 0.f, 0.f, 0.f, 0.f, 0.f, 0.f, 0.f};
  v8f acc[2][4];
#pragma unroll
  for (int i = 0; i < 2; ++i)
#pragma unroll
    for (int j = 0; j < 4; ++j) acc[i][j] = vzero;

  const bf16*  Abf = (const bf16*)Aptr;
  const float* Af  = (const float*)Aptr;
  const bf16*  Bbf = (const bf16*)Bptr;
  const float* Bf  = (const float*)Bptr;

  // staging map: 256 threads cover 128 rows x 32 k as (row = tid>>1, 16-elem half)
  const int srow = tid >> 1;
  const int skc  = (tid & 1) * 16;

  constexpr bool ASYNC_A = !AF32;
  constexpr bool ASYNC_B = (!BF32) && (!NGUARD);

  for (int k0 = 0; k0 < Kdim; k0 += KSTEP) {
    __syncthreads();
    // ---- stage A tile: As[m][k] ----
    if constexpr (AF32) {
      const float* src = Af + (size_t)(m0 + srow) * lda + k0 + skc;
#pragma unroll
      for (int j = 0; j < 16; ++j) As[srow * LDSS + skc + j] = (bf16)src[j];
    } else {
      async_lds_b128x2(Abf + (size_t)(m0 + srow) * lda + k0 + skc,
                       &As[srow * LDSS + skc]);
    }
    // ---- stage B tile (n-major): Bs[n][k] = Bsrc[n*ldb + k] ----
    if constexpr (BF32) {
      bool inb = !NGUARD || (n0 + srow) < Ndim;
      const float* src = Bf + (size_t)(n0 + srow) * ldb + k0 + skc;
#pragma unroll
      for (int j = 0; j < 16; ++j)
        Bs[srow * LDSS + skc + j] = inb ? (bf16)src[j] : (bf16)0.0f;
    } else if constexpr (!NGUARD) {
      async_lds_b128x2(Bbf + (size_t)(n0 + srow) * ldb + k0 + skc,
                       &Bs[srow * LDSS + skc]);
    } else {
      bool inb = (n0 + srow) < Ndim;
      bf16* dst = &Bs[srow * LDSS + skc];
      if (inb) {
        const uint4* s4 = (const uint4*)(Bbf + (size_t)(n0 + srow) * ldb + k0 + skc);
        *(uint4*)dst       = s4[0];
        *(uint4*)(dst + 8) = s4[1];
      } else {
        const uint4 z4 = {0u, 0u, 0u, 0u};
        *(uint4*)dst       = z4;
        *(uint4*)(dst + 8) = z4;
      }
    }
#if USE_ASYNC_LDS
    if constexpr (ASYNC_A || ASYNC_B) wait_asynccnt0();
#endif
    __syncthreads();

    if (k0 + KSTEP < Kdim) {
      if constexpr (AF32)
        __builtin_prefetch(Af + (size_t)(m0 + srow) * lda + k0 + KSTEP, 0, 0);
      else
        __builtin_prefetch(Abf + (size_t)(m0 + srow) * lda + k0 + KSTEP, 0, 0);
    }

    // ---- fragments per ISA 16-bit layouts ----
    // A 16x32: lane<16 -> M=lane, K {0..7,16..23}; lane>=16 -> K {8..15,24..31}
    // B 32x16 (n-major in LDS): lane -> N=lane&15, lane<16 -> K {0..7,16..23}; lane>=16 -> K {8..15,24..31}
    const int arow = lane & 15;
    const int akb  = (lane >> 4) * 8;
    const int bn   = lane & 15;
    const int bkb  = (lane >> 4) * 8;

    FragBf a[2], b[4];
#pragma unroll
    for (int i = 0; i < 2; ++i) {
      int r = wm + i * 16 + arow;
      a[i].u[0] = *(const uint4*)&As[r * LDSS + akb];
      a[i].u[1] = *(const uint4*)&As[r * LDSS + akb + 16];
    }
#pragma unroll
    for (int j = 0; j < 4; ++j) {
      int c = wn + j * 16 + bn;
      b[j].u[0] = *(const uint4*)&Bs[c * LDSS + bkb];
      b[j].u[1] = *(const uint4*)&Bs[c * LDSS + bkb + 16];
    }
#pragma unroll
    for (int i = 0; i < 2; ++i)
#pragma unroll
      for (int j = 0; j < 4; ++j)
        acc[i][j] = __builtin_amdgcn_wmma_f32_16x16x32_bf16(
            false, a[i].v, false, b[j].v, (short)0, acc[i][j], false, false);
  }

  // ---- epilogue; C 16x16 layout: row=(lane>>4)*8+r, col=lane&15 ----
  const int rbase = (lane >> 4) * 8;
  const int cbase = lane & 15;

  float msc = 0.0f, gam = 0.0f;
  if constexpr (EPI == 4) {
    msc = softplusf(rawm[0]) + 0.001f;
    gam = softplusf(rawg[0]);
  }

#pragma unroll
  for (int i = 0; i < 2; ++i) {
#pragma unroll
    for (int j = 0; j < 4; ++j) {
#pragma unroll
      for (int r = 0; r < 8; ++r) {
        int row = m0 + wm + i * 16 + rbase + r;
        int col = n0 + wn + j * 16 + cbase;
        float val = acc[i][j][r];
        if constexpr (EPI == 0) {
          if (!NGUARD || col < Ndim) Cout[(size_t)row * ldc + col] = val;
        } else if constexpr (EPI == 1) {
          float xx = val + bias[col];
          size_t idx = (size_t)row * Ndim + col;
          OutBf[idx] = (bf16)geluf(xx);
          Gout[idx]  = (bf16)gelugf(xx);
        } else if constexpr (EPI == 2) {
          float xx = val + bias[col];
          size_t idx = (size_t)row * Ndim + col;
          OutBf[idx] = (bf16)(W3[col] * gelugf(xx));
        } else if constexpr (EPI == 3) {
          size_t idx = (size_t)row * Ndim + col;
          OutBf[idx] = (bf16)(val * (float)Gin[idx]);
        } else {  // EPI == 4: grad -> velocity/position update (DT = 1)
          size_t idx = (size_t)row * Ndim + col;
          float vv = (Vbuf[idx] - val / msc) / (1.0f + gam);
          Vbuf[idx] = vv;
          Hbuf[idx] += vv;
        }
      }
    }
  }
}

// ------------------------------------------------------------------
extern "C" void kernel_launch(void* const* d_in, const int* in_sizes, int n_in,
                              void* d_out, int out_size, void* d_ws, size_t ws_size,
                              hipStream_t stream) {
  (void)in_sizes; (void)n_in; (void)out_size; (void)ws_size;
  const int*   x    = (const int*)d_in[0];
  const float* E    = (const float*)d_in[1];
  const float* P    = (const float*)d_in[2];
  const float* W0   = (const float*)d_in[3];
  const float* b0   = (const float*)d_in[4];
  const float* W1   = (const float*)d_in[5];
  const float* b1   = (const float*)d_in[6];
  const float* W2   = (const float*)d_in[7];
  const float* b2   = (const float*)d_in[8];
  const float* W3   = (const float*)d_in[9];
  // d_in[10] = b3 (constant offset, no effect on gradient -> unused)
  const float* rawm = (const float*)d_in[11];
  const float* rawg = (const float*)d_in[12];
  float* logits = (float*)d_out;

  // ---- workspace carve-up ----
  char* ws = (char*)d_ws;
  size_t off = 0;
  auto alloc = [&](size_t bytes) -> void* {
    void* p = ws + off;
    off = (off + bytes + 255) & ~(size_t)255;
    return p;
  };
  float* h    = (float*)alloc((size_t)M_ * D_ * 4);
  float* v    = (float*)alloc((size_t)M_ * D_ * 4);
  float* csum = (float*)alloc((size_t)B_ * NC_ * D_ * 4);
  bf16*  z    = (bf16*)alloc((size_t)M_ * 2 * D_ * 2);
  bf16*  z1   = (bf16*)alloc((size_t)M_ * H_ * 2);
  bf16*  g0   = (bf16*)alloc((size_t)M_ * H_ * 2);
  bf16*  z2   = (bf16*)alloc((size_t)M_ * H_ * 2);
  bf16*  g1   = (bf16*)alloc((size_t)M_ * H_ * 2);
  bf16*  da2  = (bf16*)alloc((size_t)M_ * H_ * 2);
  bf16*  da1  = (bf16*)alloc((size_t)M_ * H_ * 2);
  bf16*  da0  = (bf16*)alloc((size_t)M_ * H_ * 2);
  bf16*  W0b  = (bf16*)alloc((size_t)2 * D_ * H_ * 2);  // straight (backward)
  bf16*  W1b  = (bf16*)alloc((size_t)H_ * H_ * 2);
  bf16*  W2b  = (bf16*)alloc((size_t)H_ * H_ * 2);
  bf16*  W0t  = (bf16*)alloc((size_t)2 * D_ * H_ * 2);  // transposed (forward)
  bf16*  W1t  = (bf16*)alloc((size_t)H_ * H_ * 2);
  bf16*  W2t  = (bf16*)alloc((size_t)H_ * H_ * 2);

  // ---- 1) embedding + v=0 ----
  embed_kernel<<<M_, D_, 0, stream>>>(x, E, P, h, v);

  // ---- 2) weights to bf16: straight + transposed copies ----
  {
    int n0c = 2 * D_ * H_, n1c = H_ * H_;
    cvt_bf16_kernel<<<(n0c + 255) / 256, 256, 0, stream>>>(W0, W0b, n0c);
    cvt_bf16_kernel<<<(n1c + 255) / 256, 256, 0, stream>>>(W1, W1b, n1c);
    cvt_bf16_kernel<<<(n1c + 255) / 256, 256, 0, stream>>>(W2, W2b, n1c);
    cvt_t_bf16_kernel<<<(n0c + 255) / 256, 256, 0, stream>>>(W0, W0t, 2 * D_, H_);
    cvt_t_bf16_kernel<<<(n1c + 255) / 256, 256, 0, stream>>>(W1, W1t, H_, H_);
    cvt_t_bf16_kernel<<<(n1c + 255) / 256, 256, 0, stream>>>(W2, W2t, H_, H_);
  }

  const dim3 blk(256);
  const dim3 grdH(H_ / TILE, M_ / TILE);                 // (4, 32)
  const dim3 grdD(D_ / TILE, M_ / TILE);                 // (1, 32)
  const dim3 grdV((V_ + TILE - 1) / TILE, M_ / TILE);    // (393, 32)
  const dim3 grdScan(B_, NC_);

  // ---- 3) L force-field iterations ----
  for (int l = 0; l < L_; ++l) {
    chunk_sum_kernel<<<grdScan, D_, 0, stream>>>(h, csum);
    pack_kernel<<<grdScan, D_, 0, stream>>>(h, csum, z);
    // forward: a0 = z@W0+b0 -> z1=gelu, g0=gelu'   (B = W0t, ldb=2D)
    gemm_wmma<1, false, false, false><<<grdH, blk, 0, stream>>>(
        z, W0t, M_, H_, 2 * D_, 2 * D_, 2 * D_, 0, b0, nullptr, z1, nullptr, g0,
        nullptr, nullptr, nullptr, nullptr, nullptr);
    // a1 = z1@W1+b1 -> z2=gelu, g1=gelu'           (B = W1t, ldb=H)
    gemm_wmma<1, false, false, false><<<grdH, blk, 0, stream>>>(
        z1, W1t, M_, H_, H_, H_, H_, 0, b1, nullptr, z2, nullptr, g1,
        nullptr, nullptr, nullptr, nullptr, nullptr);
    // a2 = z2@W2+b2 -> da2 = W3 * gelu'(a2)        (B = W2t)
    gemm_wmma<2, false, false, false><<<grdH, blk, 0, stream>>>(
        z2, W2t, M_, H_, H_, H_, H_, 0, b2, W3, da2, nullptr, nullptr,
        nullptr, nullptr, nullptr, nullptr, nullptr);
    // da1 = (da2 @ W2^T) * g1                      (B = W2b straight)
    gemm_wmma<3, false, false, false><<<grdH, blk, 0, stream>>>(
        da2, W2b, M_, H_, H_, H_, H_, 0, nullptr, nullptr, da1, g1, nullptr,
        nullptr, nullptr, nullptr, nullptr, nullptr);
    // da0 = (da1 @ W1^T) * g0                      (B = W1b straight)
    gemm_wmma<3, false, false, false><<<grdH, blk, 0, stream>>>(
        da1, W1b, M_, H_, H_, H_, H_, 0, nullptr, nullptr, da0, g0, nullptr,
        nullptr, nullptr, nullptr, nullptr, nullptr);
    // grad_h = da0 @ (W0[D:2D,:])^T ; fused v/h update
    gemm_wmma<4, false, false, false><<<grdD, blk, 0, stream>>>(
        da0, W0b + (size_t)D_ * H_, M_, D_, H_, H_, H_, 0, nullptr, nullptr,
        nullptr, nullptr, nullptr, nullptr, h, v, rawm, rawg);
  }

  // ---- 4) logits = h @ E^T  (fp32 inputs converted on the fly) ----
  gemm_wmma<0, true, true, true><<<grdV, blk, 0, stream>>>(
      h, E, M_, V_, D_, D_, D_, V_, nullptr, nullptr, nullptr, nullptr,
      nullptr, logits, nullptr, nullptr, nullptr, nullptr);
}
